// ScatterMoE_64450279244285
// MI455X (gfx1250) — matrix operand order repaired and checked by reading, not compiled, not run on hardware
//
#include <hip/hip_runtime.h>
#include <math.h>

#if defined(__has_builtin)
#if __has_builtin(__builtin_amdgcn_global_load_async_to_lds_b128) && \
    __has_builtin(__builtin_amdgcn_s_wait_asynccnt)
#define USE_ASYNC_LDS 1
#endif
#endif
#ifndef USE_ASYNC_LDS
#define USE_ASYNC_LDS 0
#endif

namespace {

constexpr int kT = 4096;    // tokens = B*S
constexpr int kH = 2048;    // hidden
constexpr int kI = 4096;    // intermediate
constexpr int kE = 8;       // experts
constexpr int kCap = 8192;  // worst-case assignments per expert (= T*K)

constexpr int BM = 128, BN = 128, BK = 64;
constexpr int LDA = BK + 8;  // 72 elems -> 144B row stride (16B aligned, bank-skewed)
constexpr int LDB = BK + 8;

typedef __bf16 bf16_t;
typedef bf16_t v16bf __attribute__((ext_vector_type(16)));
typedef float v8f __attribute__((ext_vector_type(8)));

union Frag {
  uint4 u[2];
  v16bf v;
};

static __device__ __forceinline__ unsigned short f2bf(float f) {
  union { float f; unsigned int u; } cv;
  cv.f = f;
  unsigned int u = cv.u;
  u += 0x7FFFu + ((u >> 16) & 1u);  // round-to-nearest-even
  return (unsigned short)(u >> 16);
}

#if USE_ASYNC_LDS
typedef int v4i_vec __attribute__((vector_size(16)));  // matches builtin param type
#define ASYNC_LOAD_B128(gp, lp)                                             \
  __builtin_amdgcn_global_load_async_to_lds_b128(                           \
      (__attribute__((address_space(1))) v4i_vec*)(void*)(gp),              \
      (__attribute__((address_space(3))) v4i_vec*)(void*)(lp), 0, 0)
#endif

// ---------------- fp32 -> bf16 conversion (vectorized, layout-preserving) ----------------
__global__ __launch_bounds__(256) void cvt_bf16_kernel(const float* __restrict__ in,
                                                       unsigned short* __restrict__ out,
                                                       long n4) {
  const long i = (long)blockIdx.x * 256 + threadIdx.x;
  if (i >= n4) return;
  const float4 v = ((const float4*)in)[i];
  const unsigned int lo = (unsigned int)f2bf(v.x) | ((unsigned int)f2bf(v.y) << 16);
  const unsigned int hi = (unsigned int)f2bf(v.z) | ((unsigned int)f2bf(v.w) << 16);
  ((uint2*)out)[i] = make_uint2(lo, hi);
}

// ------- fp32 [E][K][N] -> bf16 [E][N][K] transposed conversion (LDS-tiled, coalesced) -------
__global__ __launch_bounds__(256) void cvt_transpose_kernel(const float* __restrict__ in,
                                                            unsigned short* __restrict__ out,
                                                            int Kdim, int Ntot) {
  __shared__ float tile[32][33];
  const int e = blockIdx.z;
  const int n0 = blockIdx.x * 32;
  const int k0 = blockIdx.y * 32;
  const int tx = threadIdx.x & 31;
  const int ty = threadIdx.x >> 5;  // 0..7
  const float* src = in + (long)e * Kdim * Ntot;
  unsigned short* dst = out + (long)e * Ntot * Kdim;
#pragma unroll
  for (int i = 0; i < 4; ++i)
    tile[ty + 8 * i][tx] = src[(long)(k0 + ty + 8 * i) * Ntot + n0 + tx];
  __syncthreads();
#pragma unroll
  for (int i = 0; i < 4; ++i)
    dst[(long)(n0 + ty + 8 * i) * Kdim + k0 + tx] = f2bf(tile[tx][ty + 8 * i]);
}

// ---------------- router: logits, softmax, top-2, assignment ----------------
__global__ __launch_bounds__(256) void router_kernel(const float* __restrict__ x,
                                                     const float* __restrict__ gate,
                                                     float* __restrict__ logits_out,
                                                     int* __restrict__ counts,
                                                     int* __restrict__ meta,
                                                     float* __restrict__ wgt) {
  const int t = blockIdx.x * 8 + (threadIdx.x >> 5);  // one wave32 per token
  const int lane = threadIdx.x & 31;
  float acc[kE];
#pragma unroll
  for (int e = 0; e < kE; ++e) acc[e] = 0.f;
  const float* xr = x + (long)t * kH;
  for (int j = lane; j < kH; j += 32) {
    const float xv = xr[j];
#pragma unroll
    for (int e = 0; e < kE; ++e) acc[e] += xv * gate[e * kH + j];
  }
#pragma unroll
  for (int e = 0; e < kE; ++e) {
    float s = acc[e];
#pragma unroll
    for (int o = 16; o > 0; o >>= 1) s += __shfl_xor(s, o, 32);
    acc[e] = s;
  }
  if (lane == 0) {
    float mx = acc[0];
#pragma unroll
    for (int e = 1; e < kE; ++e) mx = fmaxf(mx, acc[e]);
    float p[kE];
    float sum = 0.f;
#pragma unroll
    for (int e = 0; e < kE; ++e) {
      p[e] = expf(acc[e] - mx);
      sum += p[e];
    }
    const float inv = 1.f / sum;
#pragma unroll
    for (int e = 0; e < kE; ++e) {
      p[e] *= inv;
      logits_out[t * kE + e] = acc[e];
    }
    int i1 = 0;
#pragma unroll
    for (int e = 1; e < kE; ++e)
      if (p[e] > p[i1]) i1 = e;  // strict > : lowest index wins ties (lax.top_k)
    int i2 = (i1 == 0) ? 1 : 0;
#pragma unroll
    for (int e = 0; e < kE; ++e)
      if (e != i1 && p[e] > p[i2]) i2 = e;
    const int s1 = atomicAdd(&counts[i1], 1);
    meta[i1 * kCap + s1] = (t << 1);
    wgt[i1 * kCap + s1] = p[i1];
    const int s2 = atomicAdd(&counts[i2], 1);
    meta[i2 * kCap + s2] = (t << 1) | 1;
    wgt[i2 * kCap + s2] = p[i2];
  }
}

// ---------------- tiny exclusive scan of expert counts ----------------
__global__ void scan_kernel(const int* __restrict__ counts, int* __restrict__ offs) {
  if (threadIdx.x == 0) {
    int a = 0;
#pragma unroll
    for (int e = 0; e < kE; ++e) {
      offs[e] = a;
      a += counts[e];
    }
    offs[kE] = a;
  }
}

// ---------------- grouped (gathered) GEMM with WMMA bf16 ----------------
// B weights are pre-transposed: [E][Ntot][Kdim], K contiguous per output column.
// EPI==0: h = gelu(x[token] @ w_fc[e])  -> bf16 h_out rows (compact offs[e]+slot)
// EPI==1: y[(t<<1)|k] = w * (h @ w_proj[e]) -> fp32 y_out
template <int EPI>
__global__ __launch_bounds__(256) void moe_gemm_kernel(
    const unsigned short* __restrict__ Aglob,  // bf16 bits: x [T,H] or h [T*K, I]
    const unsigned short* __restrict__ BglobT, // bf16 bits: [E, Ntot, Kdim]
    const int* __restrict__ counts, const int* __restrict__ offs,
    const int* __restrict__ meta, const float* __restrict__ wgt,
    unsigned short* __restrict__ h_out, float* __restrict__ y_out,
    int Kdim, int Ntot) {
  const int e = blockIdx.z;
  const int cnt = counts[e];
  const int m0 = blockIdx.y * BM;
  if (m0 >= cnt) return;  // block-uniform early exit
  const int n0 = blockIdx.x * BN;
  const int ofs = offs[e];

#if USE_ASYNC_LDS
  __shared__ unsigned short As[2][BM * LDA];  // ping-pong, [row m][k] K-contiguous
  __shared__ unsigned short Bs[2][BN * LDB];  // ping-pong, [col n][k] K-contiguous
#else
  __shared__ unsigned short As[1][BM * LDA];
  __shared__ unsigned short Bs[1][BN * LDB];
#endif

  const int tid = threadIdx.x;
  const int lane = tid & 31;
  const int wave = tid >> 5;
  const int wm = wave & 3;   // 4 waves along M (32 rows each)
  const int wn = wave >> 2;  // 2 waves along N (64 cols each)
  const int lg = lane >> 4;  // lane group (0: lanes 0-15, 1: lanes 16-31)
  const int ll = lane & 15;

  // Gathered A-row base addresses; out-of-range rows clamp to a valid row
  // (their C rows are discarded by the guarded epilogue -> staging is branch-free).
  long arow[4];
#pragma unroll
  for (int it = 0; it < 4; ++it) {
    const int r = (tid >> 3) + 32 * it;  // 0..127 tile row
    int slot = m0 + r;
    slot = (slot < cnt) ? slot : (cnt - 1);
    if (EPI == 0) {
      arow[it] = (long)(meta[e * kCap + slot] >> 1) * Kdim;  // token row in x
    } else {
      arow[it] = (long)(ofs + slot) * Kdim;  // compact h row
    }
  }
  const int aseg = tid & 7;        // 8 x uint4 per 64-elem row
  const int bcol = tid >> 3;       // base col for B staging (pattern repeats +32)

  const unsigned short* Be = BglobT + (long)e * Kdim * Ntot;

  v8f acc[2][4];
#pragma unroll
  for (int mi = 0; mi < 2; ++mi)
#pragma unroll
    for (int ni = 0; ni < 4; ++ni)
#pragma unroll
      for (int r = 0; r < 8; ++r) acc[mi][ni][r] = 0.f;

  // --- compute one 64-K stage from LDS buffers ---
  auto compute_stage = [&](const unsigned short* Asb, const unsigned short* Bsb) {
#pragma unroll
    for (int kk = 0; kk < BK; kk += 32) {
      // A 16x32 fragment: lanes 0-15 row M=ll hold K{0..7} in V0..3, K{16..23} in V4..7;
      //                   lanes 16-31 hold K{8..15}, K{24..31}.
      Frag a[2], b[4];
#pragma unroll
      for (int mi = 0; mi < 2; ++mi) {
        const unsigned short* base = &Asb[(wm * 32 + mi * 16 + ll) * LDA + kk + lg * 8];
        a[mi].u[0] = *(const uint4*)(base);
        a[mi].u[1] = *(const uint4*)(base + 16);
      }
      // B 32x16 fragment: lane = col N; lanes 0-15 hold K{0..7},K{16..23}; lanes 16-31 K{8..15},K{24..31}.
#pragma unroll
      for (int ni = 0; ni < 4; ++ni) {
        const unsigned short* base = &Bsb[(wn * 64 + ni * 16 + ll) * LDB + kk + lg * 8];
        b[ni].u[0] = *(const uint4*)(base);
        b[ni].u[1] = *(const uint4*)(base + 16);
      }
#pragma unroll
      for (int mi = 0; mi < 2; ++mi)
#pragma unroll
        for (int ni = 0; ni < 4; ++ni)
          acc[mi][ni] = __builtin_amdgcn_wmma_f32_16x16x32_bf16(
              false, a[mi].v, false, b[ni].v, (short)0, acc[mi][ni], false, false);
    }
  };

#if USE_ASYNC_LDS
  // --- double-buffered async (global -> LDS DMA, ASYNCcnt-tracked) pipeline ---
  auto issue_stage = [&](int k0, int buf) {
#pragma unroll
    for (int it = 0; it < 4; ++it) {
      const int r = (tid >> 3) + 32 * it;
      ASYNC_LOAD_B128(Aglob + arow[it] + k0 + aseg * 8, &As[buf][r * LDA + aseg * 8]);
    }
#pragma unroll
    for (int it = 0; it < 4; ++it) {
      const int col = bcol + 32 * it;
      ASYNC_LOAD_B128(Be + (long)(n0 + col) * Kdim + k0 + aseg * 8,
                      &Bs[buf][col * LDB + aseg * 8]);
    }
  };

  const int nstages = Kdim / BK;
  issue_stage(0, 0);
  for (int s = 0; s < nstages; ++s) {
    if (s + 1 < nstages) {
      issue_stage((s + 1) * BK, (s + 1) & 1);
      __builtin_amdgcn_s_wait_asynccnt(8);  // in-order: stage s's 8 ops complete
    } else {
      __builtin_amdgcn_s_wait_asynccnt(0);
    }
    __syncthreads();  // all waves' stage-s data visible
    compute_stage(&As[s & 1][0], &Bs[s & 1][0]);
    __syncthreads();  // buffer s&1 free before stage s+2 overwrites it
  }
#else
  // --- fallback: single-buffer staging through VGPRs ---
  for (int k0 = 0; k0 < Kdim; k0 += BK) {
    __syncthreads();
#pragma unroll
    for (int it = 0; it < 4; ++it) {
      const int r = (tid >> 3) + 32 * it;
      const uint4 val = *(const uint4*)(Aglob + arow[it] + k0 + aseg * 8);
      *(uint4*)(&As[0][r * LDA + aseg * 8]) = val;
    }
#pragma unroll
    for (int it = 0; it < 4; ++it) {
      const int col = bcol + 32 * it;
      const uint4 val = *(const uint4*)(Be + (long)(n0 + col) * Kdim + k0 + aseg * 8);
      *(uint4*)(&Bs[0][col * LDB + aseg * 8]) = val;
    }
    if (k0 + BK < Kdim) {
      const long knext = k0 + BK;
#pragma unroll
      for (int it = 0; it < 4; ++it)
        __builtin_prefetch((const void*)(Aglob + arow[it] + knext + aseg * 8), 0, 3);
#pragma unroll
      for (int it = 0; it < 4; ++it)
        __builtin_prefetch(
            (const void*)(Be + (long)(n0 + bcol + 32 * it) * Kdim + knext + aseg * 8), 0, 3);
    }
    __syncthreads();
    compute_stage(&As[0][0], &Bs[0][0]);
  }
#endif

  // Epilogue. C layout: lanes 0-15 -> (M=r, N=ll); lanes 16-31 -> (M=r+8, N=ll).
#pragma unroll
  for (int mi = 0; mi < 2; ++mi) {
#pragma unroll
    for (int r = 0; r < 8; ++r) {
      const int m_in = wm * 32 + mi * 16 + lg * 8 + r;
      const int slot = m0 + m_in;
      if (slot >= cnt) continue;
      if (EPI == 0) {
        const long orow = (long)(ofs + slot) * Ntot + n0;
#pragma unroll
        for (int ni = 0; ni < 4; ++ni) {
          const float v = acc[mi][ni][r];
          const float g = 0.5f * v * (1.0f + erff(v * 0.70710678118654752f));  // exact gelu
          h_out[orow + wn * 64 + ni * 16 + ll] = f2bf(g);
        }
      } else {
        const int mk = meta[e * kCap + slot];  // (t<<1)|k
        const float w = wgt[e * kCap + slot];
        const long orow = (long)mk * Ntot + n0;
#pragma unroll
        for (int ni = 0; ni < 4; ++ni) {
          y_out[orow + wn * 64 + ni * 16 + ll] = w * acc[mi][ni][r];
        }
      }
    }
  }
}

// ---------------- deterministic top-2 combine ----------------
__global__ __launch_bounds__(256) void combine_kernel(const float* __restrict__ y,
                                                      float* __restrict__ out) {
  const long i = (long)blockIdx.x * 256 + threadIdx.x;  // over T*H/4
  const long flat = i * 4;
  const long t = flat / kH;
  const long c = flat - t * kH;
  const float* r0 = y + t * 2 * kH + c;
  const float* r1 = r0 + kH;
  const float4 a = *(const float4*)r0;
  const float4 b = *(const float4*)r1;
  float4 o;
  o.x = a.x + b.x;
  o.y = a.y + b.y;
  o.z = a.z + b.z;
  o.w = a.w + b.w;
  *(float4*)(out + flat) = o;
}

}  // namespace

extern "C" void kernel_launch(void* const* d_in, const int* in_sizes, int n_in,
                              void* d_out, int out_size, void* d_ws, size_t ws_size,
                              hipStream_t stream) {
  (void)in_sizes; (void)n_in; (void)out_size; (void)ws_size;
  const float* x = (const float*)d_in[0];      // [T, H]
  const float* gate = (const float*)d_in[1];   // [E, H]
  const float* wfc = (const float*)d_in[2];    // [E, H, I]
  const float* wproj = (const float*)d_in[3];  // [E, I, H]
  float* out = (float*)d_out;                  // final [T*H] then logits [T*E]
  float* logits_out = out + (long)kT * kH;

  char* ws = (char*)d_ws;
  size_t off = 0;
  auto take = [&](size_t bytes) -> char* {
    char* p = ws + off;
    off = (off + bytes + 255) & ~(size_t)255;
    return p;
  };
  unsigned short* x_bf = (unsigned short*)take((size_t)kT * kH * 2);           // 16 MB
  unsigned short* wfcT_bf = (unsigned short*)take((size_t)kE * kH * kI * 2);   // 128 MB [E][I][H]
  unsigned short* wprojT_bf = (unsigned short*)take((size_t)kE * kI * kH * 2); // 128 MB [E][H][I]
  unsigned short* h_buf = (unsigned short*)take((size_t)kT * 2 * kI * 2);      // 64 MB
  float* y_buf = (float*)take((size_t)kT * 2 * kH * 4);                        // 64 MB
  int* counts = (int*)take(kE * sizeof(int));
  int* offs = (int*)take((kE + 1) * sizeof(int));
  int* meta = (int*)take((size_t)kE * kCap * sizeof(int));
  float* wgt = (float*)take((size_t)kE * kCap * sizeof(float));

  (void)hipMemsetAsync(counts, 0, kE * sizeof(int), stream);

  {
    const long n4 = (long)kT * kH / 4;
    cvt_bf16_kernel<<<(unsigned)(n4 / 256), 256, 0, stream>>>(x, x_bf, n4);
  }
  // Weights: convert + transpose to [E][N][K] so GEMM B-tiles stage as pure b128 traffic.
  {
    dim3 g1(kI / 32, kH / 32, kE);  // w_fc: K=H, N=I
    cvt_transpose_kernel<<<g1, 256, 0, stream>>>(wfc, wfcT_bf, kH, kI);
    dim3 g2(kH / 32, kI / 32, kE);  // w_proj: K=I, N=H
    cvt_transpose_kernel<<<g2, 256, 0, stream>>>(wproj, wprojT_bf, kI, kH);
  }

  router_kernel<<<kT / 8, 256, 0, stream>>>(x, gate, logits_out, counts, meta, wgt);
  scan_kernel<<<1, 32, 0, stream>>>(counts, offs);

  dim3 gridA(kI / BN, kCap / BM, kE);  // (32, 64, 8)
  moe_gemm_kernel<0><<<gridA, 256, 0, stream>>>(x_bf, wfcT_bf, counts, offs, meta, wgt,
                                                h_buf, nullptr, kH, kI);
  dim3 gridB(kH / BN, kCap / BM, kE);  // (16, 64, 8)
  moe_gemm_kernel<1><<<gridB, 256, 0, stream>>>(h_buf, wprojT_bf, counts, offs, meta, wgt,
                                                nullptr, y_buf, kI, kH);

  combine_kernel<<<(unsigned)((long)kT * kH / 4 / 256), 256, 0, stream>>>(y_buf, out);
}
